// CBF_24172075942523
// MI455X (gfx1250) — matrix operand order repaired and checked by reading, not compiled, not run on hardware
//
#include <hip/hip_runtime.h>
#include <math.h>

// ---------------------------------------------------------------------------
// GCN-style CBF network on MI455X (gfx1250, wave32, WMMA).
//
//   A_hat == mask (0/1, diag incl).  A_norm @ M = dinv .* (A_hat @ (dinv .* M))
//   Heavy GEMMs (8192x8192 @ 8192x64) run on v_wmma_f32_16x16x32_f16 with f32
//   accumulation (0/1 exact in f16).
//
//   Fast path (ws >= ~139MB): mask kernel emits a f16 adjacency copy (total
//   HBM traffic 640 MB vs 768 MB), the f32 mask output is stored non-temporal
//   so the 128 MB f16 adjacency stays resident in the 192 MB L2 across both
//   GEMM layers, and the GEMM is split-K=4 (2048 waves, ~2 waves/SIMD for
//   latency hiding).  The GEMM body is deliberately single-buffered: explicit
//   double-buffering provoked acc v_mov_b64 shuffles + WMMA->VALU hazard
//   v_nops in the hot loop (seen in round-3 codegen), which cost more issue
//   slots than the partial waits saved.
//   Fallback path (small ws): GEMM reads the f32 mask from d_out directly.
// ---------------------------------------------------------------------------

#define NN     8192
#define HID    64
#define SPLITK 4

typedef __attribute__((ext_vector_type(16))) _Float16 v16h;
typedef __attribute__((ext_vector_type(8)))  float    v8f;

// ---------------------------------------------------------------------------
// Kernel 1: pairwise mask (float 0/1) -> d_out, optional f16 copy -> ws,
// row degree -> dinv = rsqrt(deg).  One wave per row; positions in LDS (64KB).
// On the fast path the f32 mask is a write-only stream -> non-temporal stores
// (keep L2 for the f16 adjacency, which both GEMM layers re-read).
// ---------------------------------------------------------------------------
template <bool WRITE_F16>
__global__ __launch_bounds__(256) void k_mask_dinv(const float* __restrict__ states,
                                                   float* __restrict__ maskout,
                                                   _Float16* __restrict__ a16,
                                                   float* __restrict__ dinv)
{
    __shared__ float2 spos[NN];
    const float2* p2 = (const float2*)states;          // states row = 4 floats; xy = elem 2*j
    for (int j = threadIdx.x; j < NN; j += 256) spos[j] = p2[2 * j];
    __syncthreads();

    const int wave = threadIdx.x >> 5;
    const int lane = threadIdx.x & 31;
    const int i    = blockIdx.x * 8 + wave;
    const float2 pi = spos[i];
    float*    row   = maskout + (size_t)i * NN;
    _Float16* row16 = WRITE_F16 ? (a16 + (size_t)i * NN) : nullptr;

    int cnt = 0;
    for (int j0 = 0; j0 < NN; j0 += 32) {
        const int j = j0 + lane;
        const float dx = pi.x - spos[j].x;
        const float dy = pi.y - spos[j].y;
        const bool m = (dx * dx + dy * dy) <= 1.0f;    // includes j == i
        const float mv = m ? 1.0f : 0.0f;
        if (WRITE_F16) {
            __builtin_nontemporal_store(mv, &row[j]);  // write-only stream: TH=NT
            row16[j] = (_Float16)mv;                   // L2-resident GEMM operand
        } else {
            row[j] = mv;                               // fallback re-reads this
        }
        cnt += (int)m;
    }
    for (int off = 16; off > 0; off >>= 1) cnt += __shfl_xor(cnt, off, 32);
    if (lane == 0) dinv[i] = rsqrtf((float)cnt);
}

// ---------------------------------------------------------------------------
// Kernel 2: Yt[n][i] = f16( dinv[i] * (states[i] @ W1)[n] )   (transposed!)
// Transposed layout makes the WMMA B-operand a contiguous 32-B load per lane.
// ---------------------------------------------------------------------------
__global__ __launch_bounds__(256) void k_y1t(const float* __restrict__ states,
                                             const float* __restrict__ W1,
                                             const float* __restrict__ dinv,
                                             _Float16* __restrict__ Yt)
{
    __shared__ float w1s[4 * HID];
    if (threadIdx.x < 4 * HID) w1s[threadIdx.x] = W1[threadIdx.x];
    __syncthreads();

    const int i = blockIdx.x * 256 + threadIdx.x;
    const float4 s = ((const float4*)states)[i];
    const float di = dinv[i];
#pragma unroll
    for (int n = 0; n < HID; ++n) {
        float x = s.x * w1s[n] + s.y * w1s[HID + n] + s.z * w1s[2 * HID + n] + s.w * w1s[3 * HID + n];
        Yt[(size_t)n * NN + i] = (_Float16)(di * x);   // coalesced along i per n
    }
}

// ---------------------------------------------------------------------------
// WMMA fragment conventions (wave32):
//  A (16x32 f16): lane(m=l&15, hl=l>>4) holds K runs [16hl..16hl+7], [16hl+8..+15]
//  B (32x16 f16): lane(n=l&15, kh=l>>4) holds K run [16kh..16kh+15] of col n
//  C/D (16x16 f32): lane holds col n=l&15, rows r+8*hl (r = acc element)
// ---------------------------------------------------------------------------

// Fast path: A already f16, split-K=4, raw partial tiles out (no epilogue).
__global__ __launch_bounds__(256) void k_gemm_f16(const _Float16* __restrict__ A16,
                                                  const _Float16* __restrict__ Bt,
                                                  float* __restrict__ P)   // [SPLITK][NN][HID]
{
    const int wave = threadIdx.x >> 5;
    const int lane = threadIdx.x & 31;
    const int row0 = (blockIdx.x * 8 + wave) * 16;
    const int kc   = blockIdx.y;                  // k-chunk
    const int kbeg = kc * (NN / SPLITK);
    const int kend = kbeg + (NN / SPLITK);
    const int m  = lane & 15;
    const int hl = lane >> 4;

    const _Float16* alo = A16 + (size_t)(row0 + m) * NN + 16 * hl;
    const _Float16* ahi = alo + 8;
    const _Float16* bp0 = Bt + (size_t)(m)      * NN + 16 * hl;
    const _Float16* bp1 = Bt + (size_t)(16 + m) * NN + 16 * hl;
    const _Float16* bp2 = Bt + (size_t)(32 + m) * NN + 16 * hl;
    const _Float16* bp3 = Bt + (size_t)(48 + m) * NN + 16 * hl;

    v8f acc0 = {}, acc1 = {}, acc2 = {}, acc3 = {};

    for (int k0 = kbeg; k0 < kend; k0 += 32) {
        v16h a;
        *(float4*)&a         = *(const float4*)(alo + k0);   // 8 f16 (K run 1)
        *(((float4*)&a) + 1) = *(const float4*)(ahi + k0);   // 8 f16 (K run 2)
        const v16h b0 = *(const v16h*)(bp0 + k0);
        const v16h b1 = *(const v16h*)(bp1 + k0);
        const v16h b2 = *(const v16h*)(bp2 + k0);
        const v16h b3 = *(const v16h*)(bp3 + k0);

        acc0 = __builtin_amdgcn_wmma_f32_16x16x32_f16(false, a, false, b0, (short)0, acc0, false, false);
        acc1 = __builtin_amdgcn_wmma_f32_16x16x32_f16(false, a, false, b1, (short)0, acc1, false, false);
        acc2 = __builtin_amdgcn_wmma_f32_16x16x32_f16(false, a, false, b2, (short)0, acc2, false, false);
        acc3 = __builtin_amdgcn_wmma_f32_16x16x32_f16(false, a, false, b3, (short)0, acc3, false, false);
    }

    float* pbase = P + (size_t)kc * NN * HID;
    const int n = lane & 15;
#pragma unroll
    for (int r = 0; r < 8; ++r) {
        float* prow = pbase + (size_t)(row0 + r + 8 * hl) * HID;
        prow[n]      = acc0[r];
        prow[16 + n] = acc1[r];
        prow[32 + n] = acc2[r];
        prow[48 + n] = acc3[r];
    }
}

// Reduce split-K partials + fused epilogue: H = relu(dinv .* Z + bias)
__global__ __launch_bounds__(256) void k_reduce_epi(const float* __restrict__ P,
                                                    const float* __restrict__ dinv,
                                                    const float* __restrict__ bias,
                                                    float* __restrict__ H)
{
    const int idx = blockIdx.x * 256 + threadIdx.x;     // over NN*HID
    const int i = idx >> 6;
    const int n = idx & (HID - 1);
    float z = P[idx];
#pragma unroll
    for (int c = 1; c < SPLITK; ++c) z += P[(size_t)c * NN * HID + idx];
    const float h = dinv[i] * z + bias[n];
    H[idx] = h > 0.0f ? h : 0.0f;
}

// Fallback path: A read as f32 from the mask buffer, fused epilogue, no split-K.
__global__ __launch_bounds__(256) void k_gemm_f32(const float* __restrict__ Af,
                                                  const _Float16* __restrict__ Bt,
                                                  const float* __restrict__ dinv,
                                                  const float* __restrict__ bias,
                                                  float* __restrict__ H)
{
    const int wave = threadIdx.x >> 5;
    const int lane = threadIdx.x & 31;
    const int row0 = (blockIdx.x * 8 + wave) * 16;
    const int m  = lane & 15;
    const int hl = lane >> 4;

    const float* arow = Af + (size_t)(row0 + m) * NN;
    const float* alo  = arow + 16 * hl;
    const float* ahi  = arow + 16 * hl + 8;
    const _Float16* bp0 = Bt + (size_t)(m)      * NN + 16 * hl;
    const _Float16* bp1 = Bt + (size_t)(16 + m) * NN + 16 * hl;
    const _Float16* bp2 = Bt + (size_t)(32 + m) * NN + 16 * hl;
    const _Float16* bp3 = Bt + (size_t)(48 + m) * NN + 16 * hl;

    v8f acc0 = {}, acc1 = {}, acc2 = {}, acc3 = {};

    for (int k0 = 0; k0 < NN; k0 += 32) {
        const v8f a_lo = *(const v8f*)(alo + k0);
        const v8f a_hi = *(const v8f*)(ahi + k0);
        v16h a;
#pragma unroll
        for (int e = 0; e < 8; ++e) {
            a[e]     = (_Float16)a_lo[e];
            a[e + 8] = (_Float16)a_hi[e];
        }
        const v16h b0 = *(const v16h*)(bp0 + k0);
        const v16h b1 = *(const v16h*)(bp1 + k0);
        const v16h b2 = *(const v16h*)(bp2 + k0);
        const v16h b3 = *(const v16h*)(bp3 + k0);

        acc0 = __builtin_amdgcn_wmma_f32_16x16x32_f16(false, a, false, b0, (short)0, acc0, false, false);
        acc1 = __builtin_amdgcn_wmma_f32_16x16x32_f16(false, a, false, b1, (short)0, acc1, false, false);
        acc2 = __builtin_amdgcn_wmma_f32_16x16x32_f16(false, a, false, b2, (short)0, acc2, false, false);
        acc3 = __builtin_amdgcn_wmma_f32_16x16x32_f16(false, a, false, b3, (short)0, acc3, false, false);
    }

    const int n = lane & 15;
    const float bb0 = bias[n], bb1 = bias[16 + n], bb2 = bias[32 + n], bb3 = bias[48 + n];
#pragma unroll
    for (int r = 0; r < 8; ++r) {
        const int rr = row0 + r + 8 * hl;
        const float di = dinv[rr];
        float* hrow = H + (size_t)rr * HID;
        float h0 = di * acc0[r] + bb0;  hrow[n]      = h0 > 0.0f ? h0 : 0.0f;
        float h1 = di * acc1[r] + bb1;  hrow[16 + n] = h1 > 0.0f ? h1 : 0.0f;
        float h2 = di * acc2[r] + bb2;  hrow[32 + n] = h2 > 0.0f ? h2 : 0.0f;
        float h3 = di * acc3[r] + bb3;  hrow[48 + n] = h3 > 0.0f ? h3 : 0.0f;
    }
}

// ---------------------------------------------------------------------------
// Kernel 4: Y2t[n][i] = f16( dinv[i] * (H1[i] @ W2)[n] )   (W2 staged in LDS)
// ---------------------------------------------------------------------------
__global__ __launch_bounds__(256) void k_y2t(const float* __restrict__ H1,
                                             const float* __restrict__ W2,
                                             const float* __restrict__ dinv,
                                             _Float16* __restrict__ Yt)
{
    __shared__ float w2s[HID * HID];
    for (int t = threadIdx.x; t < HID * HID; t += 256) w2s[t] = W2[t];
    __syncthreads();

    const int i = blockIdx.x * 256 + threadIdx.x;
    float h[HID];
#pragma unroll
    for (int k = 0; k < HID; ++k) h[k] = H1[(size_t)i * HID + k];
    const float di = dinv[i];
    for (int n = 0; n < HID; ++n) {
        float t = 0.0f;
#pragma unroll
        for (int k = 0; k < HID; ++k) t = fmaf(h[k], w2s[k * HID + n], t);
        Yt[(size_t)n * NN + i] = (_Float16)(di * t);
    }
}

// ---------------------------------------------------------------------------
// Kernel 5: out[i] = H2[i] . cw + cb
// ---------------------------------------------------------------------------
__global__ __launch_bounds__(256) void k_head(const float* __restrict__ H2,
                                              const float* __restrict__ cw,
                                              const float* __restrict__ cb,
                                              float* __restrict__ out)
{
    __shared__ float cws[HID];
    if (threadIdx.x < HID) cws[threadIdx.x] = cw[threadIdx.x];
    __syncthreads();

    const int i = blockIdx.x * 256 + threadIdx.x;
    float s = cb[0];
#pragma unroll
    for (int k = 0; k < HID; ++k) s = fmaf(H2[(size_t)i * HID + k], cws[k], s);
    out[i] = s;
}

// ---------------------------------------------------------------------------
extern "C" void kernel_launch(void* const* d_in, const int* in_sizes, int n_in,
                              void* d_out, int out_size, void* d_ws, size_t ws_size,
                              hipStream_t stream)
{
    const float* states = (const float*)d_in[0];
    const float* W1     = (const float*)d_in[1];
    const float* b1     = (const float*)d_in[2];
    const float* W2     = (const float*)d_in[3];
    const float* b2     = (const float*)d_in[4];
    const float* cw     = (const float*)d_in[5];
    const float* cb     = (const float*)d_in[6];

    float* out     = (float*)d_out;       // N floats
    float* maskout = out + NN;            // N*N floats (f32 adjacency)

    // Common workspace: dinv (32KB) | Yt (1MB f16) | H (2MB f32)
    char* ws = (char*)d_ws;
    size_t off = 0;
    float*    dinv = (float*)(ws + off);     off += (size_t)NN * sizeof(float);
    _Float16* Yt   = (_Float16*)(ws + off);  off += (size_t)HID * NN * sizeof(_Float16);
    float*    H    = (float*)(ws + off);     off += (size_t)NN * HID * sizeof(float);
    // Fast-path extras: split-K partials (8MB) | f16 adjacency (128MB)
    float*    P    = (float*)(ws + off);     off += (size_t)SPLITK * NN * HID * sizeof(float);
    _Float16* A16  = (_Float16*)(ws + off);  off += (size_t)NN * NN * sizeof(_Float16);

    const bool fast = (ws_size >= off);
    const dim3 blk(256);

    if (fast) {
        k_mask_dinv<true><<<NN / 8, blk, 0, stream>>>(states, maskout, A16, dinv);
        k_y1t<<<NN / 256, blk, 0, stream>>>(states, W1, dinv, Yt);

        k_gemm_f16<<<dim3(NN / 128, SPLITK), blk, 0, stream>>>(A16, Yt, P);
        k_reduce_epi<<<NN * HID / 256, blk, 0, stream>>>(P, dinv, b1, H);     // -> H1

        k_y2t<<<NN / 256, blk, 0, stream>>>(H, W2, dinv, Yt);                 // Yt reused as Y2t
        k_gemm_f16<<<dim3(NN / 128, SPLITK), blk, 0, stream>>>(A16, Yt, P);
        k_reduce_epi<<<NN * HID / 256, blk, 0, stream>>>(P, dinv, b2, H);     // -> H2
    } else {
        k_mask_dinv<false><<<NN / 8, blk, 0, stream>>>(states, maskout, nullptr, dinv);
        k_y1t<<<NN / 256, blk, 0, stream>>>(states, W1, dinv, Yt);
        k_gemm_f32<<<NN / 128, blk, 0, stream>>>(maskout, Yt, dinv, b1, H);   // -> H1
        k_y2t<<<NN / 256, blk, 0, stream>>>(H, W2, dinv, Yt);
        k_gemm_f32<<<NN / 128, blk, 0, stream>>>(maskout, Yt, dinv, b2, H);   // -> H2
    }
    k_head<<<NN / 256, blk, 0, stream>>>(H, cw, cb, out);
}
